// build_cnn16_63204738728210
// MI455X (gfx1250) — matrix operand order repaired and checked by reading, not compiled, not run on hardware
//
#include <hip/hip_runtime.h>

typedef __attribute__((ext_vector_type(16))) _Float16 v16h;
typedef __attribute__((ext_vector_type(8)))  float    v8f;
typedef __attribute__((ext_vector_type(4)))  float    f4;

#define BN_EPS 1e-5f

// Fused: conv16x16s16(3->16) + bias + BN + ReLU + 1x1(16->4) + BN + ReLU
//        + 1x1(4->1) + nearest 16x upsample.
// One wave handles a tile of 16 consecutive patches (one M=16 WMMA tile).
// Tiles: 32768 patches / 16 = 2048.  4 waves/block -> 512 blocks.
__global__ __launch_bounds__(128) void cnn16_fused_wmma(
    const float* __restrict__ in,    // (32,3,512,512)
    const float* __restrict__ w1,    // (16,3,16,16) == (16,768) row-major
    const float* __restrict__ b1,
    const float* __restrict__ g1, const float* __restrict__ beta1,
    const float* __restrict__ m1, const float* __restrict__ v1,
    const float* __restrict__ w2,    // (4,16)
    const float* __restrict__ b2,
    const float* __restrict__ g2, const float* __restrict__ beta2,
    const float* __restrict__ m2, const float* __restrict__ v2,
    const float* __restrict__ w3,    // (1,4)
    const float* __restrict__ b3,
    float* __restrict__ out)         // (32,1,512,512)
{
    __shared__ float lds_relu[4][16][16];   // [wave][patch][channel], 4 KB

    const int lane = threadIdx.x & 31;
    const int wave = threadIdx.x >> 5;
    const int tile = blockIdx.x * 4 + wave;       // 0..2047

    const int P0  = tile * 16;                    // first patch of tile
    const int b   = P0 >> 10;                     // batch   (1024 patches/img)
    const int rem = P0 & 1023;
    const int ph  = rem >> 5;                     // patch row 0..31
    const int pw0 = rem & 31;                     // 0 or 16 (tile stays in-row)

    const int hi = lane >> 4;                     // lane half (K-group select)
    const int lm = lane & 15;                     // A: M=patch, B: N=channel

    // ---- A addressing: lane lm = patch index, hi selects ix 0..7 vs 8..15
    const int x0  = (pw0 + lm) * 16 + hi * 8;     // global x of this lane's run
    const int y0g = ph * 16;                      // global y of patch top row

    // ---- B addressing: lane lm = output channel n; K contiguous in OIHW
    const float* __restrict__ wrow = w1 + lm * 768;

    v8f acc = {0.f, 0.f, 0.f, 0.f, 0.f, 0.f, 0.f, 0.f};

    // K = 768 = 3 ch * 16 rows * 16 cols; chunk of 32 = two rows of one channel
    // Unroll 4 so the compiler can keep ~4 chunks of b128 loads in flight
    // ahead of the WMMAs (low occupancy -> ILP must hide HBM latency).
#pragma unroll 4
    for (int kc = 0; kc < 768; kc += 32) {
        const int c  = kc >> 8;                   // input channel 0..2
        const int y0 = (kc & 255) >> 4;           // row pair base: 0,2,..,14

        const float* __restrict__ arow =
            in + (((size_t)(b * 3 + c) * 512 + (size_t)(y0g + y0)) * 512 + x0);
        // Input is streamed exactly once: non-temporal loads keep it from
        // evicting the L2-resident weight tensor.
        f4 a0 = __builtin_nontemporal_load((const f4*)(arow));
        f4 a1 = __builtin_nontemporal_load((const f4*)(arow + 4));
        f4 a2 = __builtin_nontemporal_load((const f4*)(arow + 512));
        f4 a3 = __builtin_nontemporal_load((const f4*)(arow + 516));

        const float* __restrict__ brow = wrow + kc + hi * 8;    // 8 at k=8h, 8 at k=16+8h
        f4 w0  = *(const f4*)(brow);              // weights: RT, stay cached
        f4 wv1 = *(const f4*)(brow + 4);
        f4 wv2 = *(const f4*)(brow + 16);
        f4 wv3 = *(const f4*)(brow + 20);

        v16h av, bv;
#pragma unroll
        for (int j = 0; j < 4; ++j) {
            av[j]      = (_Float16)a0[j];
            av[4 + j]  = (_Float16)a1[j];
            av[8 + j]  = (_Float16)a2[j];    // K+16..: second y row
            av[12 + j] = (_Float16)a3[j];
            bv[j]      = (_Float16)w0[j];
            bv[4 + j]  = (_Float16)wv1[j];
            bv[8 + j]  = (_Float16)wv2[j];
            bv[12 + j] = (_Float16)wv3[j];
        }
        acc = __builtin_amdgcn_wmma_f32_16x16x32_f16(
                  false, av, false, bv, (short)0, acc, false, false);
    }

    // ---- conv1 bias + BN1 + ReLU; D layout: lane holds channel n=lm for
    //      patches M = r + 8*hi  (r = accumulator VGPR index)
    {
        const int n  = lm;
        const float inv  = g1[n] * rsqrtf(v1[n] + BN_EPS);
        const float add  = beta1[n] - m1[n] * inv;
        const float bias = b1[n];
#pragma unroll
        for (int r = 0; r < 8; ++r) {
            const int patch = r + 8 * hi;
            const float val = (acc[r] + bias) * inv + add;
            lds_relu[wave][patch][n] = fmaxf(val, 0.f);
        }
    }
    __syncthreads();

    // ---- per-patch tail: 2 lanes per patch (q = lane/2, xh = x-half)
    const int q  = lane >> 1;
    const int xh = lane & 1;
    const float* __restrict__ lr = lds_relu[wave][q];

    float y4r[4];
#pragma unroll
    for (int o = 0; o < 4; ++o) {
        float s = b2[o];
#pragma unroll
        for (int cc = 0; cc < 16; ++cc)
            s += lr[cc] * w2[o * 16 + cc];                 // uniform s_loads
        const float inv2 = g2[o] * rsqrtf(v2[o] + BN_EPS);
        s = s * inv2 + (beta2[o] - m2[o] * inv2);
        y4r[o] = fmaxf(s, 0.f);
    }
    float y1 = b3[0];
#pragma unroll
    for (int o = 0; o < 4; ++o) y1 += y4r[o] * w3[o];

    // ---- nearest 16x upsample: splat y1 over 16x16 block, coalesced b128s.
    // Output is write-once: non-temporal stores.
    const f4 splat = { y1, y1, y1, y1 };
    float* __restrict__ obase =
        out + ((size_t)(b * 512 + y0g) * 512 + (pw0 + q) * 16 + xh * 8);
#pragma unroll
    for (int dy = 0; dy < 16; ++dy) {
        __builtin_nontemporal_store(splat, (f4*)(obase + (size_t)dy * 512));
        __builtin_nontemporal_store(splat, (f4*)(obase + (size_t)dy * 512 + 4));
    }
}

extern "C" void kernel_launch(void* const* d_in, const int* in_sizes, int n_in,
                              void* d_out, int out_size, void* d_ws, size_t ws_size,
                              hipStream_t stream) {
    (void)in_sizes; (void)n_in; (void)out_size; (void)d_ws; (void)ws_size;
    const float* in    = (const float*)d_in[0];
    const float* w1    = (const float*)d_in[1];
    const float* b1    = (const float*)d_in[2];
    const float* g1    = (const float*)d_in[3];
    const float* beta1 = (const float*)d_in[4];
    const float* m1    = (const float*)d_in[5];
    const float* v1    = (const float*)d_in[6];
    const float* w2    = (const float*)d_in[7];
    const float* b2    = (const float*)d_in[8];
    const float* g2    = (const float*)d_in[9];
    const float* beta2 = (const float*)d_in[10];
    const float* m2    = (const float*)d_in[11];
    const float* v2    = (const float*)d_in[12];
    const float* w3    = (const float*)d_in[13];
    const float* b3    = (const float*)d_in[14];
    float* out = (float*)d_out;

    // 2048 M-tiles, 4 wave32 tiles per 128-thread block -> 512 blocks
    hipLaunchKernelGGL(cnn16_fused_wmma, dim3(512), dim3(128), 0, stream,
                       in, w1, b1, g1, beta1, m1, v1,
                       w2, b2, g2, beta2, m2, v2, w3, b3, out);
}
